// LiquidMalwareDetector_51393578664047
// MI455X (gfx1250) — matrix operand order repaired and checked by reading, not compiled, not run on hardware
//
#include <hip/hip_runtime.h>
#include <hip/hip_bf16.h>

typedef __attribute__((ext_vector_type(16))) __bf16 v16bf;
typedef __attribute__((ext_vector_type(8)))  __bf16 v8bf;
typedef __attribute__((ext_vector_type(8)))  float  v8f;

#define B_TOTAL 8192
#define SEQ     1024
#define HID     64
#define ROWS    32      // batch rows per block
#define HPITCH  72      // padded LDS row pitch (bf16 elems): 144B, 16B aligned, conflict-free

// ---- branch-free native transcendentals (v_exp_f32 is base-2) ----
__device__ __forceinline__ float fast_exp2(float x) { return __builtin_amdgcn_exp2f(x); }
__device__ __forceinline__ float fast_rcp(float x)  { return __builtin_amdgcn_rcpf(x); }

__device__ __forceinline__ float fast_tanh(float x) {
    // tanh(x) = sign(x) * (1 - e^{-2|x|}) / (1 + e^{-2|x|}), e^{-2|x|} = 2^{-2*log2(e)*|x|}
    float ax = __builtin_fabsf(x);
    float t  = fast_exp2(-2.88539008178f * ax);
    float r  = (1.0f - t) * fast_rcp(1.0f + t);
    return __builtin_copysignf(r, x);
}

__device__ __forceinline__ float fast_sigmoid(float z) {
    float t = fast_exp2(-1.44269504089f * z);
    return fast_rcp(1.0f + t);
}

// ---------------- kernel 1: zero workspace stats ----------------
__global__ void lm_zero_ws(float* ws) {
    if (threadIdx.x < 8) ws[threadIdx.x] = 0.0f;
}

// ---------------- kernel 2: BatchNorm stats (sum, sumsq per channel) ----------------
__global__ void lm_bn_stats(const float* __restrict__ x, float* __restrict__ ws) {
    const long long N = (long long)B_TOTAL * SEQ;
    float s0 = 0.f, s1 = 0.f, s2 = 0.f, q0 = 0.f, q1 = 0.f, q2 = 0.f;
    const long long stride = (long long)gridDim.x * blockDim.x;
    for (long long i = (long long)blockIdx.x * blockDim.x + threadIdx.x; i < N; i += stride) {
        float a = x[i * 3 + 0], b = x[i * 3 + 1], c = x[i * 3 + 2];
        s0 += a; s1 += b; s2 += c;
        q0 += a * a; q1 += b * b; q2 += c * c;
    }
    #pragma unroll
    for (int off = 16; off > 0; off >>= 1) {
        s0 += __shfl_down(s0, off, 32);
        s1 += __shfl_down(s1, off, 32);
        s2 += __shfl_down(s2, off, 32);
        q0 += __shfl_down(q0, off, 32);
        q1 += __shfl_down(q1, off, 32);
        q2 += __shfl_down(q2, off, 32);
    }
    if ((threadIdx.x & 31) == 0) {
        atomicAdd(&ws[0], s0); atomicAdd(&ws[1], s1); atomicAdd(&ws[2], s2);
        atomicAdd(&ws[3], q0); atomicAdd(&ws[4], q1); atomicAdd(&ws[5], q2);
    }
}

// ---------------- kernel 3: fused BN-affine + liquid scan + classifier ----------------
__global__ __launch_bounds__(256, 1) void lm_liquid_scan(
    const float* __restrict__ x,     const float* __restrict__ times,
    const float* __restrict__ gamma, const float* __restrict__ beta,
    const float* __restrict__ ff1_w, const float* __restrict__ ff1_b,
    const float* __restrict__ ff2_w, const float* __restrict__ ff2_b,
    const float* __restrict__ ta_w,  const float* __restrict__ ta_b,
    const float* __restrict__ tb_w,  const float* __restrict__ tb_b,
    const float* __restrict__ cls_w, const float* __restrict__ cls_b,
    const float* __restrict__ ws,    float* __restrict__ out)
{
    __shared__ __bf16 hbuf[ROWS * HPITCH];   // hidden state, bf16, padded rows
    __shared__ float  xbuf[ROWS * 4];        // per-row {xn0, xn1, xn2, ct}

    const int tid  = threadIdx.x;
    const int wave = tid >> 5;
    const int lane = tid & 31;
    const int mt   = wave >> 2;              // row tile: 0..1 (16 rows each)
    const int nt   = wave & 3;               // hidden-col tile: 0..3 (16 cols each)
    const int b0   = blockIdx.x * ROWS;
    const int hsel = (lane < 16) ? 0 : 8;    // D/C layout row-half select

    // ---- BN affine parameters from stats ----
    const float invN = 1.0f / (float)((long long)B_TOTAL * SEQ);
    float scl[3], sft[3];
    #pragma unroll
    for (int f = 0; f < 3; ++f) {
        float mean = ws[f] * invN;
        float var  = ws[3 + f] * invN - mean * mean;
        float s    = gamma[f] * rsqrtf(var + 1e-5f);
        scl[f] = s;
        sft[f] = beta[f] - mean * s;
    }

    // ---- per-lane hidden column constants (D/C layout: lane&15 -> N) ----
    const int hcol = nt * 16 + (lane & 15);
    float wx1[3], wx2[3];
    #pragma unroll
    for (int f = 0; f < 3; ++f) {
        wx1[f] = ff1_w[hcol * 67 + f];
        wx2[f] = ff2_w[hcol * 67 + f];
    }
    const float bia1 = ff1_b[hcol], bia2 = ff2_b[hcol];
    const float ga = ta_w[hcol];
    const float gb = ta_b[hcol] + tb_w[hcol];
    const float gc = tb_b[hcol];

    // ---- preload recurrent-weight B fragments (16-bit B 32x16 layout:
    //      lane n = l&15, K = kchunk*32 + (l<16 ? 0 : 8) + j (j<8), +16+(j-8) (j>=8)) ----
    v16bf w1[2], w2[2];
    {
        const int koff = (lane < 16) ? 0 : 8;
        #pragma unroll
        for (int c = 0; c < 2; ++c) {
            const float* p1 = &ff1_w[hcol * 67 + 3 + c * 32 + koff];
            const float* p2 = &ff2_w[hcol * 67 + 3 + c * 32 + koff];
            #pragma unroll
            for (int j = 0; j < 16; ++j) {
                w1[c][j] = (__bf16)p1[j + (j & 8)];
                w2[c][j] = (__bf16)p2[j + (j & 8)];
            }
        }
    }

    // ---- zero the hidden state ----
    for (int i = tid; i < ROWS * HPITCH; i += 256) hbuf[i] = (__bf16)0.0f;

    // ---- preload step-0 inputs (wave 0, lane == batch row within block) ----
    float px0 = 0.f, px1 = 0.f, px2 = 0.f, pt = 0.f;
    if (wave == 0) {
        const float* xp = x + (long long)(b0 + lane) * SEQ * 3;
        px0 = xp[0]; px1 = xp[1]; px2 = xp[2];
        pt  = times[(long long)(b0 + lane) * SEQ];
    }
    __syncthreads();

    // A-fragment base address (16-bit A 16x32 layout: lane m = l&15,
    // halves 0..7 -> K base+0..7, halves 8..15 -> K base+16..23; base = (l<16?0:8))
    const int abase = (mt * 16 + (lane & 15)) * HPITCH + hsel;
    const int rbase = mt * 16 + hsel;

    for (int s = 0; s < SEQ; ++s) {
        // commit normalized x / clipped t for this step
        if (wave == 0) {
            xbuf[lane * 4 + 0] = px0 * scl[0] + sft[0];
            xbuf[lane * 4 + 1] = px1 * scl[1] + sft[1];
            xbuf[lane * 4 + 2] = px2 * scl[2] + sft[2];
            float ct = pt;
            ct = ct < 0.0f ? 0.0f : (ct > 60.0f ? 60.0f : ct);
            xbuf[lane * 4 + 3] = ct;
        }

        // A fragments of h (previous step's state — safe: written before last barrier)
        v16bf a0, a1;
        {
            union { v16bf v; v8bf h[2]; } u;
            u.h[0] = *(const v8bf*)&hbuf[abase +  0];
            u.h[1] = *(const v8bf*)&hbuf[abase + 16];
            a0 = u.v;
            u.h[0] = *(const v8bf*)&hbuf[abase + 32];
            u.h[1] = *(const v8bf*)&hbuf[abase + 48];
            a1 = u.v;
        }
        __syncthreads();   // xn/ct visible; all A-loads landed in regs

        // prefetch next step's global inputs (hidden behind WMMA + VALU)
        if (wave == 0 && s + 1 < SEQ) {
            const float* xp = x + ((long long)(b0 + lane) * SEQ + (s + 1)) * 3;
            px0 = xp[0]; px1 = xp[1]; px2 = xp[2];
            pt  = times[(long long)(b0 + lane) * SEQ + (s + 1)];
        }

        // C-init = bias + x_t @ Wx^T (folded input contribution);
        // gate g depends only on ct -> compute BEFORE WMMA so trans latency overlaps
        v8f c1, c2;
        float garr[8];
        #pragma unroll
        for (int r = 0; r < 8; ++r) {
            float4 xc = *(const float4*)&xbuf[(rbase + r) * 4];
            c1[r] = bia1 + xc.x * wx1[0] + xc.y * wx1[1] + xc.z * wx1[2];
            c2[r] = bia2 + xc.x * wx2[0] + xc.y * wx2[1] + xc.z * wx2[2];
            float ct = xc.w;
            garr[r] = fast_sigmoid((ga * ct + gb) * ct + gc);
        }

        // recurrent h @ Wh^T via WMMA, K=64 in 2 chunks, f32 accumulate
        c1 = __builtin_amdgcn_wmma_f32_16x16x32_bf16(false, a0, false, w1[0], (short)0, c1, false, false);
        c1 = __builtin_amdgcn_wmma_f32_16x16x32_bf16(false, a1, false, w1[1], (short)0, c1, false, false);
        c2 = __builtin_amdgcn_wmma_f32_16x16x32_bf16(false, a0, false, w2[0], (short)0, c2, false, false);
        c2 = __builtin_amdgcn_wmma_f32_16x16x32_bf16(false, a1, false, w2[1], (short)0, c2, false, false);

        // branch-free tanh mix, write new h back to LDS as bf16
        #pragma unroll
        for (int r = 0; r < 8; ++r) {
            float t1 = fast_tanh(c1[r]);
            float t2 = fast_tanh(c2[r]);
            float nh = t2 + garr[r] * (t1 - t2);
            hbuf[(rbase + r) * HPITCH + hcol] = (__bf16)nh;
        }
        __syncthreads();   // new h visible for next step
    }

    // ---- classifier + softmax (one lane per batch row) ----
    if (wave == 0) {
        float l0 = cls_b[0], l1 = cls_b[1];
        #pragma unroll 4
        for (int h = 0; h < HID; ++h) {
            float hv = (float)hbuf[lane * HPITCH + h];
            l0 += hv * cls_w[h];
            l1 += hv * cls_w[HID + h];
        }
        float mx = fmaxf(l0, l1);
        float e0 = fast_exp2(1.44269504089f * (l0 - mx));
        float e1 = fast_exp2(1.44269504089f * (l1 - mx));
        float inv = fast_rcp(e0 + e1);
        out[(long long)(b0 + lane) * 2 + 0] = e0 * inv;
        out[(long long)(b0 + lane) * 2 + 1] = e1 * inv;
    }
}

extern "C" void kernel_launch(void* const* d_in, const int* in_sizes, int n_in,
                              void* d_out, int out_size, void* d_ws, size_t ws_size,
                              hipStream_t stream)
{
    const float* x     = (const float*)d_in[0];
    const float* times = (const float*)d_in[1];
    const float* gamma = (const float*)d_in[2];
    const float* beta  = (const float*)d_in[3];
    const float* ff1w  = (const float*)d_in[4];
    const float* ff1b  = (const float*)d_in[5];
    const float* ff2w  = (const float*)d_in[6];
    const float* ff2b  = (const float*)d_in[7];
    const float* taw   = (const float*)d_in[8];
    const float* tab   = (const float*)d_in[9];
    const float* tbw   = (const float*)d_in[10];
    const float* tbb   = (const float*)d_in[11];
    const float* clsw  = (const float*)d_in[12];
    const float* clsb  = (const float*)d_in[13];
    float* out = (float*)d_out;
    float* ws  = (float*)d_ws;

    lm_zero_ws<<<1, 32, 0, stream>>>(ws);
    lm_bn_stats<<<1024, 256, 0, stream>>>(x, ws);
    lm_liquid_scan<<<B_TOTAL / ROWS, 256, 0, stream>>>(
        x, times, gamma, beta, ff1w, ff1b, ff2w, ff2b,
        taw, tab, tbw, tbb, clsw, clsb, ws, out);
}
